// MultiHeadAttention_35356170781144
// MI455X (gfx1250) — compile-verified, matrix operand order checked by reading, not run on hardware
//
#include <hip/hip_runtime.h>

// MI455X / gfx1250 multi-head attention forward.
// Pipeline: [gemm Q][gemm KV][flash-attn][gemm Out], all bf16 WMMA f32-acc.
// Flash-attn stages K/V tiles with GLOBAL_LOAD_ASYNC_TO_LDS_B128 (ASYNCcnt,
// double-buffered) + global_prefetch, per CDNA5 ISA ch.10 async ops.
// Workspace layout (requires ws_size >= 32 MiB):
//   Qb  : bf16 [4096,1024]  (8 MiB)
//   KVb : bf16 [4096,2048]  (16 MiB)
//   Eb  : bf16 [4096,1024]  (8 MiB)

typedef __attribute__((ext_vector_type(16))) __bf16 v16bf;
typedef __attribute__((ext_vector_type(8)))  __bf16 v8bf;
typedef __attribute__((ext_vector_type(8)))  float  v8f;
typedef __attribute__((ext_vector_type(4)))  int    v4i;

#define HID   1024
#define NH    16
#define HD    64
#define SQ    2048
#define SK    2048
#define SCALE_F 0.03125f   // 1/sqrt(1024)

#if __has_builtin(__builtin_amdgcn_global_load_async_to_lds_b128)
#define HAVE_ASYNC_LDS 1
#else
#define HAVE_ASYNC_LDS 0
#endif

typedef __attribute__((address_space(1))) v4i gv4i;  // global v4i (AS1)
typedef __attribute__((address_space(3))) v4i lv4i;  // LDS v4i (AS3)

static __device__ __forceinline__ gv4i* as_global_v4(const void* p) {
  return (gv4i*)(unsigned long long)(uintptr_t)p;
}
static __device__ __forceinline__ lv4i* as_lds_v4(const void* p) {
  // generic shared pointer: low 32 bits are the LDS byte address (ISA 10.2)
  return (lv4i*)(unsigned int)(uintptr_t)p;
}

static __device__ __forceinline__ v16bf frag16(const __bf16* row, int kbase, int g) {
  // 16-bit A/B fragment (ISA 7.12.2): lane half g holds K = kbase+g*8+0..7
  // in VGPRs 0..3 and K = kbase+16+g*8+0..7 in VGPRs 4..7.
  v8bf lo = *(const v8bf*)(row + kbase + g * 8);
  v8bf hi = *(const v8bf*)(row + kbase + 16 + g * 8);
  return __builtin_shufflevector(lo, hi, 0,1,2,3,4,5,6,7,8,9,10,11,12,13,14,15);
}

static __device__ __forceinline__ v8f wmma_bf16(v16bf a, v16bf b, v8f c) {
  return __builtin_amdgcn_wmma_f32_16x16x32_bf16(false, a, false, b, (short)0, c,
                                                 false, false);
}

// ---------------------------------------------------------------------------
// Generic GEMM:  C[M,N] = A[M,K] * W[N,K]^T   (A fp32 or bf16, W fp32,
// C fp32 or bf16).  64x64 block tile, 4 waves x (16 rows x 64 cols), K step 32.
// ---------------------------------------------------------------------------
template <bool A_BF16, bool OUT_F32>
__global__ __launch_bounds__(128) void gemm_wmma(const void* __restrict__ Ain,
                                                 const float* __restrict__ W,
                                                 void* __restrict__ Cout,
                                                 int M, int N, int K) {
  __shared__ __align__(16) __bf16 sA[64 * 40];
  __shared__ __align__(16) __bf16 sW[64 * 40];

  const int tid  = threadIdx.x;
  const int wave = tid >> 5;
  const int lane = tid & 31;
  const int g    = lane >> 4;
  const int ln   = lane & 15;
  const int n0   = blockIdx.x * 64;
  const int m0   = blockIdx.y * 64;

  v8f acc[4] = {};

  const int r  = tid >> 1;        // staging row 0..63
  const int hc = (tid & 1) * 16;  // staging col half 0 / 16

  for (int kk = 0; kk < K; kk += 32) {
    __syncthreads();
    // L2 prefetch of the next K tile (global_prefetch_b8)
    if (kk + 32 < K) {
      if (A_BF16)
        __builtin_prefetch((const __bf16*)Ain + (size_t)(m0 + r) * K + kk + 32 + hc, 0, 1);
      else
        __builtin_prefetch((const float*)Ain + (size_t)(m0 + r) * K + kk + 32 + hc, 0, 1);
      __builtin_prefetch(W + (size_t)(n0 + r) * K + kk + 32 + hc, 0, 1);
    }
    // ---- stage A tile [64 x 32] as bf16 ----
    if (A_BF16) {
      const __bf16* ap = (const __bf16*)Ain + (size_t)(m0 + r) * K + kk + hc;
      *(v8bf*)&sA[r * 40 + hc]     = *(const v8bf*)ap;
      *(v8bf*)&sA[r * 40 + hc + 8] = *(const v8bf*)(ap + 8);
    } else {
      const float* ap = (const float*)Ain + (size_t)(m0 + r) * K + kk + hc;
      float4 f0 = *(const float4*)(ap + 0);
      float4 f1 = *(const float4*)(ap + 4);
      float4 f2 = *(const float4*)(ap + 8);
      float4 f3 = *(const float4*)(ap + 12);
      v8bf b0{}, b1{};
      b0[0]=(__bf16)f0.x; b0[1]=(__bf16)f0.y; b0[2]=(__bf16)f0.z; b0[3]=(__bf16)f0.w;
      b0[4]=(__bf16)f1.x; b0[5]=(__bf16)f1.y; b0[6]=(__bf16)f1.z; b0[7]=(__bf16)f1.w;
      b1[0]=(__bf16)f2.x; b1[1]=(__bf16)f2.y; b1[2]=(__bf16)f2.z; b1[3]=(__bf16)f2.w;
      b1[4]=(__bf16)f3.x; b1[5]=(__bf16)f3.y; b1[6]=(__bf16)f3.z; b1[7]=(__bf16)f3.w;
      *(v8bf*)&sA[r * 40 + hc]     = b0;
      *(v8bf*)&sA[r * 40 + hc + 8] = b1;
    }
    // ---- stage W tile [64 x 32] (rows = output cols) as bf16 ----
    {
      const float* wp = W + (size_t)(n0 + r) * K + kk + hc;
      float4 f0 = *(const float4*)(wp + 0);
      float4 f1 = *(const float4*)(wp + 4);
      float4 f2 = *(const float4*)(wp + 8);
      float4 f3 = *(const float4*)(wp + 12);
      v8bf b0{}, b1{};
      b0[0]=(__bf16)f0.x; b0[1]=(__bf16)f0.y; b0[2]=(__bf16)f0.z; b0[3]=(__bf16)f0.w;
      b0[4]=(__bf16)f1.x; b0[5]=(__bf16)f1.y; b0[6]=(__bf16)f1.z; b0[7]=(__bf16)f1.w;
      b1[0]=(__bf16)f2.x; b1[1]=(__bf16)f2.y; b1[2]=(__bf16)f2.z; b1[3]=(__bf16)f2.w;
      b1[4]=(__bf16)f3.x; b1[5]=(__bf16)f3.y; b1[6]=(__bf16)f3.z; b1[7]=(__bf16)f3.w;
      *(v8bf*)&sW[r * 40 + hc]     = b0;
      *(v8bf*)&sW[r * 40 + hc + 8] = b1;
    }
    __syncthreads();

    v16bf a = frag16(&sA[(wave * 16 + ln) * 40], 0, g);
#pragma unroll
    for (int nt = 0; nt < 4; ++nt) {
      v16bf b = frag16(&sW[(nt * 16 + ln) * 40], 0, g);
      acc[nt] = wmma_bf16(a, b, acc[nt]);
    }
  }

  // ---- epilogue: C/D layout -> lane ln = col, vgpr i = row g*8+i ----
#pragma unroll
  for (int nt = 0; nt < 4; ++nt) {
#pragma unroll
    for (int i = 0; i < 8; ++i) {
      int row = m0 + wave * 16 + g * 8 + i;
      int col = n0 + nt * 16 + ln;
      if (OUT_F32)
        ((float*)Cout)[(size_t)row * N + col] = acc[nt][i];
      else
        ((__bf16*)Cout)[(size_t)row * N + col] = (__bf16)acc[nt][i];
    }
  }
}

// ---------------------------------------------------------------------------
// Flash attention: one block per (q-tile 64, head, batch); 4 waves, each wave
// owns a 16-row q stripe; stream sk in 64-key tiles.  K/V tiles are staged
// with double-buffered GLOBAL_LOAD_ASYNC_TO_LDS_B128 (ASYNCcnt).
// ---------------------------------------------------------------------------
__global__ __launch_bounds__(128) void flash_attn_wmma(
    const __bf16* __restrict__ Qb,   // [2*2048, 1024]
    const __bf16* __restrict__ KVb,  // [2*2048, 2048]  (K | V)
    __bf16* __restrict__ Eb) {       // [2*2048, 1024]
  __shared__ __align__(16) __bf16 sK[2][64 * 72];   // K tile, row = key (ping-pong)
  __shared__ __align__(16) __bf16 sVr[2][64 * 72];  // V tile row-major (ping-pong)
  __shared__ __align__(16) __bf16 sVt[64 * 72];     // V tile transposed, row = dim
  __shared__ __align__(16) __bf16 sP[4 * 16 * 72];  // per-wave P scratch

  const int tid  = threadIdx.x;
  const int wave = tid >> 5;
  const int lane = tid & 31;
  const int g    = lane >> 4;
  const int ln   = lane & 15;
  const int qt   = blockIdx.x;   // q tile (64 rows)
  const int h    = blockIdx.y;   // head
  const int bz   = blockIdx.z;   // batch
  const int q0   = qt * 64;

  // --- load Q fragments once (A layout, rows = q, K = head dim 0..63) ---
  const __bf16* qrow =
      Qb + (size_t)(bz * SQ + q0 + wave * 16 + ln) * HID + h * HD;
  v16bf qa0 = frag16(qrow, 0, g);
  v16bf qa1 = frag16(qrow, 32, g);

  float m_i[8], l_i[8];
  v8f acc[4] = {};
#pragma unroll
  for (int i = 0; i < 8; ++i) { m_i[i] = -1e30f; l_i[i] = 0.0f; }

  __bf16* pbase = &sP[wave * 16 * 72];

  // stage one 64-key K/V tile into buffer `buf`
  auto stage = [&](int buf, int kt) {
    const __bf16* g0 = KVb + (size_t)(bz * SK + kt * 64) * (2 * HID) + h * HD;
#if HAVE_ASYNC_LDS
#pragma unroll
    for (int u = tid; u < 512; u += 128) {
      int row = u >> 3, c8 = (u & 7) * 8;
      __builtin_amdgcn_global_load_async_to_lds_b128(
          as_global_v4(g0 + (size_t)row * (2 * HID) + c8),
          as_lds_v4(&sK[buf][row * 72 + c8]), 0, 0);
      __builtin_amdgcn_global_load_async_to_lds_b128(
          as_global_v4(g0 + (size_t)row * (2 * HID) + HID + c8),
          as_lds_v4(&sVr[buf][row * 72 + c8]), 0, 0);
    }
#else
#pragma unroll
    for (int u = tid; u < 512; u += 128) {
      int row = u >> 3, c8 = (u & 7) * 8;
      *(v8bf*)&sK[buf][row * 72 + c8] =
          *(const v8bf*)(g0 + (size_t)row * (2 * HID) + c8);
      *(v8bf*)&sVr[buf][row * 72 + c8] =
          *(const v8bf*)(g0 + (size_t)row * (2 * HID) + HID + c8);
    }
#endif
  };

  stage(0, 0);

  for (int kt = 0; kt < SK / 64; ++kt) {
    const int cur = kt & 1;
#if HAVE_ASYNC_LDS
    asm volatile("s_wait_asynccnt 0" ::: "memory");  // current buffer arrived
#endif
    __syncthreads();                                 // all waves done w/ prev tile
    if (kt + 1 < SK / 64) stage(cur ^ 1, kt + 1);    // overlap next tile load
    if (kt + 2 < SK / 64) {                          // GL2 prefetch 2 tiles ahead
      const __bf16* pf = KVb +
          (size_t)(bz * SK + (kt + 2) * 64 + (tid >> 1)) * (2 * HID) + h * HD +
          (tid & 1) * HID;
      __builtin_prefetch((const void*)pf, 0, 1);
    }
    // ---- LDS->LDS transpose of V tile (B fragment wants K-major keys) ----
#pragma unroll
    for (int u = tid; u < 512; u += 128) {
      int row = u >> 3, c8 = (u & 7) * 8;
      v8bf vv = *(const v8bf*)&sVr[cur][row * 72 + c8];
#pragma unroll
      for (int e = 0; e < 8; ++e) sVt[(c8 + e) * 72 + row] = vv[e];
    }
    __syncthreads();

    // ---- S = Q K^T  (4 key subtiles of 16, 2 k-steps of 32 over head dim) --
    v8f s[4] = {};
#pragma unroll
    for (int nt = 0; nt < 4; ++nt) {
      const __bf16* krow = &sK[cur][(nt * 16 + ln) * 72];
      s[nt] = wmma_bf16(qa0, frag16(krow, 0, g), s[nt]);
      s[nt] = wmma_bf16(qa1, frag16(krow, 32, g), s[nt]);
    }

    // ---- online softmax: row stats live entirely inside a 16-lane half ----
#pragma unroll
    for (int i = 0; i < 8; ++i) {
      float t = fmaxf(fmaxf(s[0][i], s[1][i]), fmaxf(s[2][i], s[3][i]));
#pragma unroll
      for (int mask = 1; mask < 16; mask <<= 1)
        t = fmaxf(t, __shfl_xor(t, mask, 16));
      float mn = fmaxf(m_i[i], t * SCALE_F);
      float al = __expf(m_i[i] - mn);
      m_i[i] = mn;
      l_i[i] *= al;
#pragma unroll
      for (int nt = 0; nt < 4; ++nt) acc[nt][i] *= al;
    }

    // ---- P = exp(S*scale - m); C-layout -> A-layout via per-wave LDS ----
    float rs[8] = {};
#pragma unroll
    for (int nt = 0; nt < 4; ++nt) {
#pragma unroll
      for (int i = 0; i < 8; ++i) {
        float p = __expf(s[nt][i] * SCALE_F - m_i[i]);
        rs[i] += p;
        pbase[(g * 8 + i) * 72 + nt * 16 + ln] = (__bf16)p;
      }
    }
#pragma unroll
    for (int i = 0; i < 8; ++i) {
      float rsum = rs[i];
#pragma unroll
      for (int mask = 1; mask < 16; mask <<= 1)
        rsum += __shfl_xor(rsum, mask, 16);
      l_i[i] += rsum;
    }

    // ---- O += P V  (A = P [16 x 64 keys], B = Vt rows = dim) ----
    const __bf16* prow = &pbase[ln * 72];
    v16bf pa0 = frag16(prow, 0, g);
    v16bf pa1 = frag16(prow, 32, g);
#pragma unroll
    for (int nt = 0; nt < 4; ++nt) {
      const __bf16* vrow = &sVt[(nt * 16 + ln) * 72];
      acc[nt] = wmma_bf16(pa0, frag16(vrow, 0, g), acc[nt]);
      acc[nt] = wmma_bf16(pa1, frag16(vrow, 32, g), acc[nt]);
    }
  }

  // ---- normalize and write E (bf16) ----
#pragma unroll
  for (int i = 0; i < 8; ++i) {
    float inv = 1.0f / l_i[i];
    int row = q0 + wave * 16 + g * 8 + i;
    __bf16* ep = Eb + (size_t)(bz * SQ + row) * HID + h * HD + ln;
#pragma unroll
    for (int nt = 0; nt < 4; ++nt)
      ep[nt * 16] = (__bf16)(acc[nt][i] * inv);
  }
}

// ---------------------------------------------------------------------------
extern "C" void kernel_launch(void* const* d_in, const int* in_sizes, int n_in,
                              void* d_out, int out_size, void* d_ws,
                              size_t ws_size, hipStream_t stream) {
  (void)in_sizes; (void)n_in; (void)out_size; (void)ws_size;
  const float* query   = (const float*)d_in[0];  // [2,2048,1024]
  const float* context = (const float*)d_in[1];  // [2,2048,1024]
  const float* Wq      = (const float*)d_in[2];  // [1024,1024]
  const float* Wkv     = (const float*)d_in[3];  // [2048,1024]
  const float* Wout    = (const float*)d_in[4];  // [1024,1024]
  float* out = (float*)d_out;

  __bf16* Qb  = (__bf16*)d_ws;                   // 4096x1024
  __bf16* KVb = Qb + (size_t)4096 * 1024;        // 4096x2048
  __bf16* Eb  = KVb + (size_t)4096 * 2048;       // 4096x1024

  dim3 blk(128);
  // Q projection
  gemm_wmma<false, false><<<dim3(1024 / 64, 4096 / 64), blk, 0, stream>>>(
      query, Wq, Qb, 4096, 1024, 1024);
  // KV projection
  gemm_wmma<false, false><<<dim3(2048 / 64, 4096 / 64), blk, 0, stream>>>(
      context, Wkv, KVb, 4096, 2048, 1024);
  // attention
  flash_attn_wmma<<<dim3(SQ / 64, NH, 2), blk, 0, stream>>>(Qb, KVb, Eb);
  // output projection
  gemm_wmma<true, true><<<dim3(1024 / 64, 4096 / 64), blk, 0, stream>>>(
      Eb, Wout, out, 4096, 1024, 1024);
}
